// ScaledDotProductAttention_21964462752504
// MI455X (gfx1250) — matrix operand order repaired and checked by reading, not compiled, not run on hardware
//
#include <hip/hip_runtime.h>

typedef _Float16 v16h __attribute__((ext_vector_type(16)));
typedef float    v8f  __attribute__((ext_vector_type(8)));

constexpr int D     = 64;    // head dim
constexpr int S     = 2048;  // sequence length
constexpr int BH    = 64;    // B*H
constexpr int BQ    = 128;   // query rows per workgroup
constexpr int BK    = 64;    // keys per kv step
constexpr int WAVES = 8;     // 256 threads / wave32

// scale = 1/sqrt(64), folded with log2(e) so softmax runs in base-2 (v_exp_f32)
#define QSCALE (0.125f * 1.44269504088896340736f)
#define NEGBIG (-3.0e38f)

// ---- 16-lane butterfly reduction via DPP (no LDS, no waits) ----
template <int CTRL>
__device__ __forceinline__ float dpp_xchg(float x) {
  return __builtin_bit_cast(
      float, __builtin_amdgcn_update_dpp(0, __builtin_bit_cast(int, x), CTRL,
                                         0xf, 0xf, true));
}
__device__ __forceinline__ float row16_max(float x) {
  x = fmaxf(x, dpp_xchg<0xB1>(x));   // quad_perm [1,0,3,2]
  x = fmaxf(x, dpp_xchg<0x4E>(x));   // quad_perm [2,3,0,1]
  x = fmaxf(x, dpp_xchg<0x141>(x));  // row_half_mirror
  x = fmaxf(x, dpp_xchg<0x140>(x));  // row_mirror
  return x;
}
__device__ __forceinline__ float row16_sum(float x) {
  x += dpp_xchg<0xB1>(x);
  x += dpp_xchg<0x4E>(x);
  x += dpp_xchg<0x141>(x);
  x += dpp_xchg<0x140>(x);
  return x;
}

__global__ __launch_bounds__(256)
void fa_causal_wmma_kernel(const float* __restrict__ Q,
                           const float* __restrict__ K,
                           const float* __restrict__ V,
                           float* __restrict__ Out) {
  __shared__ _Float16 Kl[2][BK * D];       // K tiles [buf][key][dim] 16 KB
  __shared__ _Float16 Vt[2][D * BK];       // V tiles [buf][dim][key] 16 KB
  __shared__ _Float16 Pl[WAVES][16 * BK];  // per-wave P [m][key]     16 KB

  const int tid  = threadIdx.x;
  const int lane = tid & 31;
  // wave id is uniform per wave: pin it to an SGPR so guards become s_cbranch
  const int wave = __builtin_amdgcn_readfirstlane(tid >> 5);
  const int ln   = lane & 15;   // N-index / M-row within fragment
  const int hg   = lane >> 4;   // half-wave group

  const int bh    = blockIdx.y;
  const int qtile = blockIdx.x;
  const int q0    = qtile * BQ + wave * 16;  // this wave's first query row

  const float* qb = Q   + (size_t)bh * S * D;
  const float* kb = K   + (size_t)bh * S * D;
  const float* vb = V   + (size_t)bh * S * D;
  float*       ob = Out + (size_t)bh * S * D;

  // cooperative loader geometry (16 contiguous floats per thread)
  const int ckey = tid >> 2;          // 0..63
  const int cdim = (tid & 3) * 16;    // 0,16,32,48

  float4 kr[4], vr[4];  // register staging for the in-flight tile
  auto load_tile = [&](int kv0) {
    const float4* kg = (const float4*)(kb + (size_t)(kv0 + ckey) * D + cdim);
    const float4* vg = (const float4*)(vb + (size_t)(kv0 + ckey) * D + cdim);
#pragma unroll
    for (int q = 0; q < 4; ++q) { kr[q] = kg[q]; vr[q] = vg[q]; }
  };
  auto store_tile = [&](int buf) {
    _Float16* kd = &Kl[buf][ckey * D + cdim];
    _Float16* vd = &Vt[buf][cdim * BK + ckey];  // transposed store
#pragma unroll
    for (int q = 0; q < 4; ++q) {
      kd[4 * q + 0] = (_Float16)kr[q].x; kd[4 * q + 1] = (_Float16)kr[q].y;
      kd[4 * q + 2] = (_Float16)kr[q].z; kd[4 * q + 3] = (_Float16)kr[q].w;
      vd[(4 * q + 0) * BK] = (_Float16)vr[q].x;
      vd[(4 * q + 1) * BK] = (_Float16)vr[q].y;
      vd[(4 * q + 2) * BK] = (_Float16)vr[q].z;
      vd[(4 * q + 3) * BK] = (_Float16)vr[q].w;
    }
  };

  // ---- Q fragments: A-layout 16x32 f16, two K-chunks over D=64 ----
  v16h qa[2];
#pragma unroll
  for (int c = 0; c < 2; ++c) {
#pragma unroll
    for (int v = 0; v < 8; ++v) {
      const int kl = 2 * (v & 3) + ((v >= 4) ? 16 : 0) + 8 * hg;
      const float* p = qb + (size_t)(q0 + ln) * D + c * 32 + kl;
      qa[c][2 * v]     = (_Float16)(p[0] * QSCALE);
      qa[c][2 * v + 1] = (_Float16)(p[1] * QSCALE);
    }
  }

  // ---- accumulators ----
  v8f   o[4];
  float mi[8], li[8];
#pragma unroll
  for (int nb = 0; nb < 4; ++nb)
#pragma unroll
    for (int r = 0; r < 8; ++r) o[nb][r] = 0.0f;
#pragma unroll
  for (int r = 0; r < 8; ++r) { mi[r] = NEGBIG; li[r] = 0.0f; }

  const int q_hi   = qtile * BQ + BQ - 1;
  const int nsteps = q_hi / BK + 1;  // causal: block-uniform kv trip count

  // ---- prologue: stage tile 0 ----
  load_tile(0);
  store_tile(0);
  __syncthreads();

  int buf = 0;
  for (int step = 0; step < nsteps; ++step) {
    const int kv0 = step * BK;

    // issue next tile's global loads now; they land during compute
    if (step + 1 < nsteps) load_tile(kv0 + BK);

    // wave-uniform (SGPR) guard: skip tiles entirely above this strip
    if (kv0 <= q0 + 15) {
      // ---- S = Q*K^T : four 16x16 N-blocks, K-depth 64 = 2 WMMAs each ----
      v8f s[4];
#pragma unroll
      for (int nb = 0; nb < 4; ++nb) {
#pragma unroll
        for (int r = 0; r < 8; ++r) s[nb][r] = 0.0f;
#pragma unroll
        for (int c = 0; c < 2; ++c) {
          v16h bf;  // B 32x16: lane -> N = ln, K = 2(v&3)(+1) + 16*(v>=4) + 8*hg
#pragma unroll
          for (int v = 0; v < 8; ++v) {
            const int kd = c * 32 + 2 * (v & 3) + ((v >= 4) ? 16 : 0) + 8 * hg;
            const _Float16* p = &Kl[buf][(nb * 16 + ln) * D + kd];
            bf[2 * v] = p[0]; bf[2 * v + 1] = p[1];
          }
          s[nb] = __builtin_amdgcn_wmma_f32_16x16x32_f16(
              false, qa[c], false, bf, (short)0, s[nb], false, false);
        }
      }

      // ---- causal mask: only the wave's diagonal tile needs it ----
      if (kv0 + BK - 1 > q0) {
#pragma unroll
        for (int nb = 0; nb < 4; ++nb)
#pragma unroll
          for (int r = 0; r < 8; ++r) {
            const int row = q0 + r + 8 * hg;     // C-layout: M = r + 8*hg
            const int key = kv0 + nb * 16 + ln;  //           N = ln
            s[nb][r] = (key <= row) ? s[nb][r] : NEGBIG;
          }
      }

      // ---- online softmax (base-2), DPP row reductions ----
      float mnew[8];
#pragma unroll
      for (int r = 0; r < 8; ++r) {
        float mx = fmaxf(fmaxf(s[0][r], s[1][r]), fmaxf(s[2][r], s[3][r]));
        mnew[r] = fmaxf(mi[r], row16_max(mx));
      }
#pragma unroll
      for (int nb = 0; nb < 4; ++nb)
#pragma unroll
        for (int r = 0; r < 8; ++r)
          s[nb][r] = __builtin_exp2f(s[nb][r] - mnew[r]);
#pragma unroll
      for (int r = 0; r < 8; ++r) {
        const float sum =
            row16_sum((s[0][r] + s[1][r]) + (s[2][r] + s[3][r]));
        const float alpha = __builtin_exp2f(mi[r] - mnew[r]);
        li[r] = li[r] * alpha + sum;
        mi[r] = mnew[r];
#pragma unroll
        for (int nb = 0; nb < 4; ++nb) o[nb][r] *= alpha;
      }

      // ---- P: C-layout f32 -> LDS -> A-layout f16 (per-wave scratch) ----
      _Float16* pw = &Pl[wave][0];
#pragma unroll
      for (int nb = 0; nb < 4; ++nb)
#pragma unroll
        for (int r = 0; r < 8; ++r)
          pw[(r + 8 * hg) * BK + nb * 16 + ln] = (_Float16)s[nb][r];

      // same-wave LDS is in-order; fence stops compiler reordering
      asm volatile("s_wait_dscnt 0" ::: "memory");

      v16h pf[2];
#pragma unroll
      for (int c = 0; c < 2; ++c)
#pragma unroll
        for (int v = 0; v < 8; ++v) {
          const int kl = 2 * (v & 3) + ((v >= 4) ? 16 : 0) + 8 * hg;
          const _Float16* p = &pw[ln * BK + c * 32 + kl];
          pf[c][2 * v] = p[0]; pf[c][2 * v + 1] = p[1];
        }

      // ---- O += P * V : 4 dim-blocks, K-depth 64 = 2 WMMAs each ----
#pragma unroll
      for (int nb = 0; nb < 4; ++nb) {
#pragma unroll
        for (int c = 0; c < 2; ++c) {
          v16h vf;
#pragma unroll
          for (int v = 0; v < 8; ++v) {
            const int key = c * 32 + 2 * (v & 3) + ((v >= 4) ? 16 : 0) + 8 * hg;
            const _Float16* p = &Vt[buf][(nb * 16 + ln) * BK + key];
            vf[2 * v] = p[0]; vf[2 * v + 1] = p[1];
          }
          o[nb] = __builtin_amdgcn_wmma_f32_16x16x32_f16(
              false, pf[c], false, vf, (short)0, o[nb], false, false);
        }
      }
    }  // active strip

    // convert + stage next tile into the other buffer, then one barrier
    if (step + 1 < nsteps) store_tile(buf ^ 1);
    __syncthreads();
    buf ^= 1;
  }  // kv loop

  // ---- epilogue: normalize by l and store fp32 ----
#pragma unroll
  for (int r = 0; r < 8; ++r) {
    const float inv = 1.0f / li[r];
    const int row = q0 + r + 8 * hg;
#pragma unroll
    for (int nb = 0; nb < 4; ++nb)
      ob[(size_t)row * D + nb * 16 + ln] = o[nb][r] * inv;
  }
}

extern "C" void kernel_launch(void* const* d_in, const int* in_sizes, int n_in,
                              void* d_out, int out_size, void* d_ws, size_t ws_size,
                              hipStream_t stream) {
  const float* q = (const float*)d_in[0];
  const float* k = (const float*)d_in[1];
  const float* v = (const float*)d_in[2];
  float* out = (float*)d_out;
  dim3 grid(S / BQ, BH);  // 16 q-tiles x 64 (batch*heads)
  fa_causal_wmma_kernel<<<grid, dim3(256), 0, stream>>>(q, k, v, out);
}
